// Recombine_Cross_Transformer_52982716563824
// MI455X (gfx1250) — matrix operand order repaired and checked by reading, not compiled, not run on hardware
//
#include <hip/hip_runtime.h>
#include <hip/hip_bf16.h>

typedef __attribute__((ext_vector_type(16))) __bf16 v16bf;
typedef __attribute__((ext_vector_type(8)))  __bf16 v8bf;
typedef __attribute__((ext_vector_type(8)))  float  v8f;
typedef __attribute__((ext_vector_type(4)))  unsigned v4u;
typedef __attribute__((ext_vector_type(8)))  unsigned v8u;
typedef unsigned short u16;

#define DIM   192
#define HEADS 8
#define CH    24          // channels per head
#define NSP   16384       // spatial (128*128)
#define BATCH 8
#define NT    128         // spatial tile per workgroup
#define KPAD  200         // padded k-stride in LDS (192 + 8)
#define NC    512         // TDM chunk width (elements) in k_gram

union Frag { v16bf v; v8bf h[2]; };

__device__ __forceinline__ u16 f2bf(float x) {
    union { float f; unsigned u; } v; v.f = x;
    unsigned r = v.u + 0x7FFFu + ((v.u >> 16) & 1u);
    return (u16)(r >> 16);
}

__device__ __forceinline__ float bf2f(u16 x) {
    union { unsigned u; float f; } v; v.u = ((unsigned)x) << 16;
    return v.f;
}

__device__ __forceinline__ unsigned lds_off(const void* p) {
    // generic shared pointer: low 32 bits are the LDS byte offset
    return (unsigned)(unsigned long long)(uintptr_t)p;
}

// Issue a TDM 2D tile load: global (row-major, row stride NSP elems, 2B elems)
// -> LDS (tile_h rows x tile_w elems, contiguous). OOB rows read as zero.
__device__ __forceinline__ void tdm_load_tile(unsigned lds_byte_off, const u16* gsrc,
                                              unsigned rows_avail, unsigned tile_w,
                                              unsigned tile_h) {
    unsigned long long ga = (unsigned long long)(uintptr_t)gsrc;
    v4u g0;
    g0[0] = 1u;                                                // count=1, user desc
    g0[1] = lds_byte_off;                                      // lds_addr
    g0[2] = (unsigned)ga;                                      // global_addr[31:0]
    g0[3] = ((unsigned)(ga >> 32) & 0x01ffffffu) | (2u << 30); // addr[56:32] | type=2
    v8u g1;
    g1[0] = 0x00010000u;                                       // data_size=1 (2 bytes)
    g1[1] = ((unsigned)NSP & 0xffffu) << 16;                   // tensor_dim0[15:0]
    g1[2] = ((unsigned)NSP >> 16) | ((rows_avail & 0xffffu) << 16); // dim0 hi | dim1 lo
    g1[3] = (rows_avail >> 16) | (tile_w << 16);               // dim1 hi | tile_dim0
    g1[4] = tile_h;                                            // tile_dim1 (tile_dim2=0)
    g1[5] = (unsigned)NSP;                                     // tensor_dim0_stride[31:0]
    g1[6] = 0u;                                                // stride hi | dim1_stride lo
    g1[7] = 0u;                                                // dim1_stride hi
    asm volatile("tensor_load_to_lds %0, %1" :: "s"(g0), "s"(g1) : "memory");
}

// ------------------------------------------------- pack weights to A-frags
// wpack layout: [w 0..7][mt 0..11][kt 0..5][lane 0..31][16 bf16]
__global__ __launch_bounds__(256) void k_pack(
    const float* w0, const float* w1, const float* w2, const float* w3,
    const float* w4, const float* w5, const float* w6, const float* w7,
    u16* wpack)
{
    int gid = blockIdx.x * 256 + threadIdx.x;   // one lane-task each
    if (gid >= 8 * 12 * 6 * 32) return;
    int lane = gid & 31;
    int kt   = (gid >> 5) % 6;
    int mt   = (gid / (32 * 6)) % 12;
    int w    = gid / (32 * 6 * 12);
    const float* Wsrc[8] = { w0, w1, w2, w3, w4, w5, w6, w7 };
    const float* W = Wsrc[w];
    int r = lane & 15, half = lane >> 4;
    int m = mt * 16 + r;
    u16* dst = wpack + (size_t)gid * 16;
    #pragma unroll
    for (int i = 0; i < 16; i++) {
        int kl = half * 16 + i;
        int k  = kt * 32 + kl;
        dst[i] = f2bf(W[m * DIM + k]);
    }
}

// ------------------------------------------------------------------- QKV
// qkv layout: [t 0..5][b][c 0..191][n 0..16383] bf16 ; t = q1,k1,v1,q2,k2,v2
__global__ __launch_bounds__(256) void k_qkv(
    const float* __restrict__ inpA, const float* __restrict__ inpB,
    const float* bA0, const float* bA1, const float* bA2,
    const float* bB0, const float* bB1, const float* bB2,
    const u16* __restrict__ wpack, u16* __restrict__ qkv)
{
    __shared__ u16 Xs[NT * KPAD];                    // k-major bf16 tile
    __shared__ float biasS[3 * DIM];
    int p = blockIdx.z, b = blockIdx.y, nb = blockIdx.x;
    int n0 = nb * NT;
    const float* inp = p ? inpB : inpA;
    const float* bv0 = p ? bB0 : bA0;
    const float* bv1 = p ? bB1 : bA1;
    const float* bv2 = p ? bB2 : bA2;
    int tid = threadIdx.x;

    // stage biases into LDS (avoid flat loads in the epilogue)
    for (int i = tid; i < 3 * DIM; i += 256)
        biasS[i] = (i < DIM) ? bv0[i] : (i < 2 * DIM) ? bv1[i - DIM] : bv2[i - 2 * DIM];

    // stage 192x128 f32 -> LDS [n][k] bf16
    for (int idx = tid; idx < DIM * NT / 4; idx += 256) {
        int c = idx >> 5;                            // row (0..191)
        int g = idx & 31;                            // float4 group
        float4 xv = *(const float4*)(inp + ((size_t)(b * DIM + c)) * NSP + n0 + g * 4);
        int nl = g * 4;
        Xs[(nl + 0) * KPAD + c] = f2bf(xv.x);
        Xs[(nl + 1) * KPAD + c] = f2bf(xv.y);
        Xs[(nl + 2) * KPAD + c] = f2bf(xv.z);
        Xs[(nl + 3) * KPAD + c] = f2bf(xv.w);
    }
    __syncthreads();

    int wave = tid >> 5, lane = tid & 31;
    int r = lane & 15, half = lane >> 4;
    int nloc = wave * 16 + r;

    // hoist the 6 B fragments (shared across all 36 output tiles)
    Frag bf[6];
    #pragma unroll
    for (int kt = 0; kt < 6; kt++) {
        const u16* bp = &Xs[nloc * KPAD + kt * 32 + half * 16];
        bf[kt].h[0] = *(const v8bf*)bp;
        bf[kt].h[1] = *(const v8bf*)(bp + 8);
    }

    int wbase = p * 3;
    for (int t3 = 0; t3 < 3; t3++) {
        const u16* wp = wpack + (size_t)(wbase + t3) * 12 * 6 * 32 * 16;
        int t = wbase + t3;
        for (int mt = 0; mt < 12; mt++) {
            v8f acc = {};
            #pragma unroll
            for (int kt = 0; kt < 6; kt++) {
                Frag af;
                af.v = *(const v16bf*)(wp + (((size_t)mt * 6 + kt) * 32 + lane) * 16);
                acc = __builtin_amdgcn_wmma_f32_16x16x32_bf16(
                          false, af.v, false, bf[kt].v, (short)0, acc, false, false);
            }
            size_t obase = ((size_t)t * BATCH + b) * DIM;
            #pragma unroll
            for (int j = 0; j < 8; j++) {
                int c = mt * 16 + half * 8 + j;
                float v = acc[j] + biasS[t3 * DIM + c];
                qkv[(obase + c) * NSP + n0 + nloc] = f2bf(v);
            }
        }
    }
}

// -------------------- norms + Gram (TDM double-buffered) + softmax + fold
// aprime layout: [p][b][h][c 0..23][d 0..23] f32
__global__ __launch_bounds__(256) void k_gram(
    const u16* __restrict__ qkv, const float* __restrict__ temp,
    float* __restrict__ aprime)
{
    __shared__ u16 Qs[2][32 * NC];     // TDM double buffers
    __shared__ u16 Ks[2][32 * NC];
    __shared__ float Gp[2][32][32];
    __shared__ float normS[72];        // [0..23]=|q|^2 [24..47]=|k|^2 [48..71]=|v|^2
    int h = blockIdx.x, b = blockIdx.y, p = blockIdx.z;
    int tq = p ? 3 : 0, tk = p ? 1 : 4, tv = p ? 2 : 5;
    int tid = threadIdx.x, wave = tid >> 5, lane = tid & 31;
    int r = lane & 15, half = lane >> 4;
    int mt = wave & 1, nt = (wave >> 1) & 1, ks = (wave >> 2) & 1;

    const int TOTROW = 6 * BATCH * DIM;
    int qrow0 = (tq * BATCH + b) * DIM + h * CH;
    int krow0 = (tk * BATCH + b) * DIM + h * CH;
    const u16* Qbase = qkv + (size_t)qrow0 * NSP;
    const u16* Kbase = qkv + (size_t)krow0 * NSP;

    // kick off first chunk DMA, overlap with norm computation
    if (wave == 0) {
        tdm_load_tile(lds_off(&Qs[0][0]), Qbase, (unsigned)(TOTROW - qrow0), NC, 32);
        tdm_load_tile(lds_off(&Ks[0][0]), Kbase, (unsigned)(TOTROW - krow0), NC, 32);
    }

    // ---- row sum-of-squares over n (9 rows per wave, lane-parallel)
    #pragma unroll
    for (int i = 0; i < 9; i++) {
        int rid = wave + i * 8;                   // 0..71
        int which = rid / CH;                     // 0=q 1=k 2=v
        int t = (which == 0) ? tq : (which == 1) ? tk : tv;
        int c = rid % CH;
        const u16* row = qkv + ((size_t)(t * BATCH + b) * DIM + h * CH + c) * NSP;
        float s = 0.0f;
        for (int x = lane * 4; x < NSP; x += 128) {
            unsigned long long pk = *(const unsigned long long*)(row + x);
            #pragma unroll
            for (int e = 0; e < 4; e++) {
                float f = bf2f((u16)(pk >> (16 * e)));
                s += f * f;
            }
        }
        s += __shfl_xor(s, 1, 32);
        s += __shfl_xor(s, 2, 32);
        s += __shfl_xor(s, 4, 32);
        s += __shfl_xor(s, 8, 32);
        s += __shfl_xor(s, 16, 32);
        if (lane == 0) normS[rid] = s;
    }

    // ---- Gram with TDM double buffering
    v8f acc = {};
    for (int ch = 0; ch < NSP / NC; ch++) {
        int buf = ch & 1;
        if (wave == 0 && ch + 1 < NSP / NC) {
            tdm_load_tile(lds_off(&Qs[buf ^ 1][0]), Qbase + (size_t)(ch + 1) * NC,
                          (unsigned)(TOTROW - qrow0), NC, 32);
            tdm_load_tile(lds_off(&Ks[buf ^ 1][0]), Kbase + (size_t)(ch + 1) * NC,
                          (unsigned)(TOTROW - krow0), NC, 32);
        }
        if (ch + 1 < NSP / NC) __builtin_amdgcn_s_wait_tensorcnt(2);
        else                   __builtin_amdgcn_s_wait_tensorcnt(0);
        __syncthreads();                           // chunk `ch` resident for all waves
        const u16* Qr = &Qs[buf][(mt * 16 + r) * NC];
        const u16* Kr = &Ks[buf][(nt * 16 + r) * NC];
        #pragma unroll
        for (int kk = 0; kk < 8; kk++) {
            int kt = ks * 8 + kk;
            Frag af, bfr;
            af.h[0]  = *(const v8bf*)(Qr + kt * 32 + half * 16);
            af.h[1]  = *(const v8bf*)(Qr + kt * 32 + half * 16 + 8);
            bfr.h[0] = *(const v8bf*)(Kr + kt * 32 + half * 16);
            bfr.h[1] = *(const v8bf*)(Kr + kt * 32 + half * 16 + 8);
            acc = __builtin_amdgcn_wmma_f32_16x16x32_bf16(
                      false, af.v, false, bfr.v, (short)0, acc, false, false);
        }
        __syncthreads();                           // done with buf before overwrite
    }
    #pragma unroll
    for (int j = 0; j < 8; j++)
        Gp[ks][mt * 16 + half * 8 + j][nt * 16 + r] = acc[j];
    __syncthreads();

    if (tid < CH) {
        int c = tid;
        float dq  = fmaxf(sqrtf(normS[c]), 1e-12f);
        float tpl = temp[h];
        float g[CH];
        float mx = -1e30f;
        for (int d = 0; d < CH; d++) {
            float dk = fmaxf(sqrtf(normS[CH + d]), 1e-12f);
            g[d] = (Gp[0][c][d] + Gp[1][c][d]) * tpl / (dq * dk);
            mx = fmaxf(mx, g[d]);
        }
        float s = 0.0f;
        for (int d = 0; d < CH; d++) { g[d] = __expf(g[d] - mx); s += g[d]; }
        float inv = 1.0f / s;
        float* ap = aprime + (((size_t)p * BATCH + b) * HEADS + h) * CH * CH + c * CH;
        for (int d = 0; d < CH; d++) {
            float dv = fmaxf(sqrtf(normS[2 * CH + d]), 1e-12f);
            ap[d] = g[d] * inv / dv;
        }
    }
}

// ---------------------------- attention apply + output projection + bias
__global__ __launch_bounds__(256) void k_apply(
    const u16* __restrict__ qkv, const float* __restrict__ aprime,
    const u16* __restrict__ wpack,
    const float* boA, const float* boB, float* __restrict__ outbuf)
{
    __shared__ u16 Vs[NT * 256];          // [n][h*32 + d'] head-padded, zeroed
    __shared__ u16 As[8 * 2 * 32 * 16];   // a' A-fragments per head (zero-padded)
    __shared__ u16 Os[NT * KPAD];         // o_pre, k-major
    __shared__ float boS[DIM];
    int p = blockIdx.z, b = blockIdx.y, nb = blockIdx.x;
    int n0 = nb * NT;
    int tv = p ? 2 : 5;
    int wo = p ? 7 : 6;
    const float* bo = p ? boB : boA;
    int tid = threadIdx.x;

    for (int i = tid; i < NT * 256; i += 256) Vs[i] = 0;
    for (int i = tid; i < DIM; i += 256) boS[i] = bo[i];
    __syncthreads();

    const u16* Vb = qkv + ((size_t)tv * BATCH + b) * DIM * NSP;
    for (int idx = tid; idx < DIM * NT; idx += 256) {
        int c  = idx >> 7;            // 0..191
        int nl = idx & 127;
        Vs[nl * 256 + (c / CH) * 32 + (c % CH)] = Vb[(size_t)c * NSP + n0 + nl];
    }

    const float* apb = aprime + ((size_t)p * BATCH + b) * HEADS * CH * CH;
    for (int idx = tid; idx < 8 * 2 * 32; idx += 256) {
        int lane = idx & 31;
        int mt   = (idx >> 5) & 1;
        int hh   = idx >> 6;
        int r = lane & 15, hf = lane >> 4;
        int m = mt * 16 + r;
        const float* ap = apb + hh * CH * CH;
        u16* dst = &As[((hh * 2 + mt) * 32 + lane) * 16];
        #pragma unroll
        for (int i = 0; i < 16; i++) {
            int kl = hf * 16 + i;
            float v = (m < CH && kl < CH) ? ap[m * CH + kl] : 0.0f;
            dst[i] = f2bf(v);
        }
    }
    __syncthreads();

    int wave = tid >> 5, lane = tid & 31;
    int r = lane & 15, half = lane >> 4;
    int nloc = wave * 16 + r;

    // GEMM1: o_pre[c'][n] = a'[c'][d] * V[d][n] per head (K=32, zero-padded)
    for (int hh = 0; hh < 8; hh++) {
        Frag bfr;
        const u16* bp = &Vs[nloc * 256 + hh * 32 + half * 16];
        bfr.h[0] = *(const v8bf*)bp;
        bfr.h[1] = *(const v8bf*)(bp + 8);
        #pragma unroll
        for (int mt = 0; mt < 2; mt++) {
            Frag af;
            af.v = *(const v16bf*)&As[((hh * 2 + mt) * 32 + lane) * 16];
            v8f acc = {};
            acc = __builtin_amdgcn_wmma_f32_16x16x32_bf16(
                      false, af.v, false, bfr.v, (short)0, acc, false, false);
            if (!(mt == 1 && half == 1)) {       // rows 24..31 are zero padding
                #pragma unroll
                for (int j = 0; j < 8; j++)
                    Os[nloc * KPAD + hh * CH + mt * 16 + half * 8 + j] = f2bf(acc[j]);
            }
        }
    }
    __syncthreads();

    // GEMM2: out = Wo * o_pre + bo
    Frag bf2[6];
    #pragma unroll
    for (int kt = 0; kt < 6; kt++) {
        const u16* bp = &Os[nloc * KPAD + kt * 32 + half * 16];
        bf2[kt].h[0] = *(const v8bf*)bp;
        bf2[kt].h[1] = *(const v8bf*)(bp + 8);
    }
    const u16* wp = wpack + (size_t)wo * 12 * 6 * 32 * 16;
    float* outp = outbuf + ((size_t)p * BATCH + b) * DIM * NSP;
    for (int mt = 0; mt < 12; mt++) {
        v8f acc = {};
        #pragma unroll
        for (int kt = 0; kt < 6; kt++) {
            Frag af;
            af.v = *(const v16bf*)(wp + (((size_t)mt * 6 + kt) * 32 + lane) * 16);
            acc = __builtin_amdgcn_wmma_f32_16x16x32_bf16(
                      false, af.v, false, bf2[kt].v, (short)0, acc, false, false);
        }
        #pragma unroll
        for (int j = 0; j < 8; j++) {
            int c = mt * 16 + half * 8 + j;
            outp[(size_t)c * NSP + n0 + nloc] = acc[j] + boS[c];
        }
    }
}

// ---------------------------------------------------------------- launcher
extern "C" void kernel_launch(void* const* d_in, const int* in_sizes, int n_in,
                              void* d_out, int out_size, void* d_ws, size_t ws_size,
                              hipStream_t stream) {
    const float* inp1 = (const float*)d_in[0];
    const float* inp2 = (const float*)d_in[1];
    const float* wq1 = (const float*)d_in[2];  const float* bq1 = (const float*)d_in[3];
    const float* wk1 = (const float*)d_in[4];  const float* bk1 = (const float*)d_in[5];
    const float* wv1 = (const float*)d_in[6];  const float* bv1 = (const float*)d_in[7];
    const float* wq2 = (const float*)d_in[8];  const float* bq2 = (const float*)d_in[9];
    const float* wk2 = (const float*)d_in[10]; const float* bk2 = (const float*)d_in[11];
    const float* wv2 = (const float*)d_in[12]; const float* bv2 = (const float*)d_in[13];
    const float* wo1 = (const float*)d_in[14]; const float* bo1 = (const float*)d_in[15];
    const float* wo2 = (const float*)d_in[16]; const float* bo2 = (const float*)d_in[17];
    const float* temp = (const float*)d_in[18];

    char* ws = (char*)d_ws;
    const size_t WPACK_OFF  = 0;                 // 8*12*6*32*16*2 = 589824 B
    const size_t APRIME_OFF = 626688;            // 2*8*8*576*4    = 294912 B
    const size_t QKV_OFF    = 1u << 20;          // 6*8*192*16384*2 bf16
    u16*   wpack  = (u16*)  (ws + WPACK_OFF);
    float* aprime = (float*)(ws + APRIME_OFF);
    u16*   qkv    = (u16*)  (ws + QKV_OFF);

    k_pack<<<(8 * 12 * 6 * 32 + 255) / 256, 256, 0, stream>>>(
        wq1, wk1, wv1, wq2, wk2, wv2, wo1, wo2, wpack);
    k_qkv<<<dim3(NSP / NT, BATCH, 2), 256, 0, stream>>>(
        inp1, inp2, bq1, bk1, bv1, bq2, bk2, bv2, wpack, qkv);
    k_gram<<<dim3(HEADS, BATCH, 2), 256, 0, stream>>>(qkv, temp, aprime);
    k_apply<<<dim3(NSP / NT, BATCH, 2), 256, 0, stream>>>(
        qkv, aprime, wpack, bo1, bo2, (float*)d_out);
}
